// ActionVQVAE_82008105550297
// MI455X (gfx1250) — compile-verified, matrix operand order checked by reading, not run on hardware
//
#include <hip/hip_runtime.h>
#include <hip/hip_bf16.h>

// ---------------------------------------------------------------------------
// ActionVQVAE forward for MI455X (gfx1250, wave32, WMMA).
// Dominant cost: B x K distance GEMM (17.2 GFLOP) -> fused bf16-WMMA + argmin,
// codebook double-buffered through LDS via async global->LDS DMA; each B tile
// amortized over two A row-tiles to stay below the LDS bandwidth ceiling.
// ---------------------------------------------------------------------------

typedef __attribute__((ext_vector_type(16))) __bf16 v16bf;
typedef __attribute__((ext_vector_type(8)))  __bf16 v8bf;
typedef __attribute__((ext_vector_type(8)))  float  v8f;

#define B_N 32768
#define A_N 6
#define H_N 256
#define D_N 128
#define K_N 2048

// Flip to 0 if the async-to-LDS inline asm is rejected by the assembler.
#ifndef VQ_ASYNC_LDS
#define VQ_ASYNC_LDS 1
#endif

// Load a 16x32 bf16 tile (row-major source, leading dim `ld`) into the
// documented wave32 A/B operand layout:
//   lanes 0-15 : row = lane,    elems 0-7 = K 0..7,  elems 8-15 = K 16..23
//   lanes 16-31: row = lane-16, elems 0-7 = K 8..15, elems 8-15 = K 24..31
// -> two contiguous 16-byte loads per lane (global_load_b128 / ds_load_b128).
__device__ __forceinline__ v16bf load_tile16x32(const __bf16* base, int ld) {
  const int l  = threadIdx.x & 31;
  const int r  = l & 15;
  const int k0 = (l < 16) ? 0 : 8;
  const __bf16* p = base + (size_t)r * (size_t)ld + k0;
  v8bf lo = *(const v8bf*)(p);
  v8bf hi = *(const v8bf*)(p + 16);
  return __builtin_shufflevector(lo, hi, 0,1,2,3,4,5,6,7,8,9,10,11,12,13,14,15);
}

// ---------------- prep kernels -------------------------------------------

// dst[n*K + k] = (bf16) src[k*N + n]   (transpose so B-operand loads are contiguous)
__global__ __launch_bounds__(256) void tconv_kernel(const float* __restrict__ src,
                                                    __bf16* __restrict__ dst,
                                                    int K, int N) {
  int g = blockIdx.x * 256 + threadIdx.x;
  if (g >= K * N) return;
  int k = g % K;
  int n = g / K;
  dst[g] = (__bf16)src[(size_t)k * N + n];
}

__global__ __launch_bounds__(256) void conv_kernel(const float* __restrict__ src,
                                                   __bf16* __restrict__ dst, int n) {
  int g = blockIdx.x * 256 + threadIdx.x;
  if (g < n) dst[g] = (__bf16)src[g];
}

// snorm[k] = 0.5 * ||emb_k||^2
__global__ __launch_bounds__(256) void snorm_kernel(const float* __restrict__ emb,
                                                    float* __restrict__ snorm) {
  int r = blockIdx.x * 256 + threadIdx.x;
  if (r >= K_N) return;
  float s = 0.f;
  const float* p = emb + (size_t)r * D_N;
  #pragma unroll 4
  for (int d = 0; d < D_N; ++d) { float v = p[d]; s += v * v; }
  snorm[r] = 0.5f * s;
}

// ---------------- encoder layer 1 (K=6, VALU) ----------------------------

__global__ __launch_bounds__(256) void enc1_kernel(const float* __restrict__ action,
                                                   const float* __restrict__ We1,
                                                   const float* __restrict__ be1,
                                                   __bf16* __restrict__ h1) {
  int g = blockIdx.x * 256 + threadIdx.x;          // B*H threads
  int row = g >> 8;
  int col = g & 255;
  float s = be1[col];
  const float* ar = action + (size_t)row * A_N;
  #pragma unroll
  for (int a = 0; a < A_N; ++a) s += ar[a] * We1[a * H_N + col];
  h1[g] = (__bf16)fmaxf(s, 0.f);
}

// ---------------- generic bf16 WMMA GEMM ---------------------------------
// out = act(A[B x KD] @ W[KD x N] + bias), W pre-transposed to [N x KD] bf16.
// Block: 256 threads = 8 waves, each wave owns one 16x16 tile -> 32x64/block.
// Whole A k-strip preloaded into VGPRs; B tile t+1 is fetched before the
// WMMA for tile t so the loads hide behind the matrix pipe.

template <int KD, bool RELU, bool WF32>
__global__ __launch_bounds__(256) void gemm_bf16_wmma(const __bf16* __restrict__ A,
                                                      const __bf16* __restrict__ Wt,
                                                      const float* __restrict__ bias,
                                                      __bf16* __restrict__ outB,
                                                      float* __restrict__ outF,
                                                      int N) {
  constexpr int T = KD / 32;
  const int wid  = threadIdx.x >> 5;
  const int lane = threadIdx.x & 31;
  const int row0 = blockIdx.x * 32 + (wid >> 2) * 16;
  const int col0 = blockIdx.y * 64 + (wid & 3) * 16;

  const __bf16* Ab = A  + (size_t)row0 * KD;
  const __bf16* Wb = Wt + (size_t)col0 * KD;

  v16bf a[T];
  #pragma unroll
  for (int t = 0; t < T; ++t) a[t] = load_tile16x32(Ab + t * 32, KD);

  v16bf b0 = load_tile16x32(Wb, KD);
  v8f c = {};
  #pragma unroll
  for (int t = 0; t < T; ++t) {
    v16bf bn;
    if (t + 1 < T) bn = load_tile16x32(Wb + (t + 1) * 32, KD);
    c = __builtin_amdgcn_wmma_f32_16x16x32_bf16(false, a[t], false, b0,
                                                (short)0, c, false, false);
    if (t + 1 < T) b0 = bn;
  }

  const int n   = lane & 15;
  const int mhi = (lane >> 4) * 8;
  const float bv = bias[col0 + n];
  #pragma unroll
  for (int v = 0; v < 8; ++v) {
    float x = c[v] + bv;
    if (RELU) x = fmaxf(x, 0.f);
    const size_t off = (size_t)(row0 + mhi + v) * N + col0 + n;
    outB[off] = (__bf16)x;
    if (WF32) outF[off] = x;
  }
}

// ---------------- fused VQ distance + argmin + gather --------------------
// Block = 4 waves x 32 rows = 128 rows (each wave owns its rows exclusively).
// score = z.e - 0.5||e||^2 (argmax == argmin of distance).
// Codebook streamed through LDS in 64-code groups (16 KB), double-buffered
// with ASYNCcnt-tracked global->LDS DMA; each 1 KB B tile feeds two WMMAs
// (row-tiles a0/a1), halving LDS B-operand bandwidth per FLOP.

#define VQ_WAVES   4
#define VQ_THREADS (VQ_WAVES * 32)
#define VQ_RPW     32                // rows per wave (2 x 16-row tiles)
#define VQ_RPB     (VQ_WAVES * VQ_RPW) // 128 rows per block
#define VQ_TG      4                 // 16-code col-tiles per LDS group
#define VQ_GCODES  (VQ_TG * 16)      // 64 codes per group
#define VQ_NG      (K_N / VQ_GCODES) // 32 groups
#define VQ_GELEMS  (VQ_GCODES * D_N) // 8192 bf16 = 16 KB per group

__device__ __forceinline__ void vq_stage_group(const __bf16* __restrict__ embb,
                                               __bf16* dst, int g) {
  const __bf16* src = embb + (size_t)g * VQ_GELEMS;
  const int tid = threadIdx.x;
  #pragma unroll
  for (int p = 0; p < VQ_GELEMS / (VQ_THREADS * 8); ++p) {  // 8 x b128 / thread
    const int c = tid + p * VQ_THREADS;                     // 16-byte chunk id
#if VQ_ASYNC_LDS
    // ASYNCcnt-tracked DMA, ISA §15.18 op 98. LDS byte offset taken from the
    // low 32 bits of the generic pointer (LDS aperture maps addr[31:0]).
    unsigned lds_off = (unsigned)(size_t)(dst + c * 8);
    unsigned long long ga = (unsigned long long)(size_t)(src + c * 8);
    asm volatile("global_load_async_to_lds_b128 %0, %1, off"
                 :: "v"(lds_off), "v"(ga) : "memory");
#else
    *(uint4*)(dst + c * 8) = *(const uint4*)(src + c * 8);
#endif
  }
}

__device__ __forceinline__ void vq_stage_wait() {
#if VQ_ASYNC_LDS
  asm volatile("s_wait_asynccnt 0x0" ::: "memory");
#endif
  __syncthreads();
}

__global__ __launch_bounds__(VQ_THREADS) void vq_kernel(const __bf16* __restrict__ zb,
                                                 const float* __restrict__ zf,
                                                 const __bf16* __restrict__ embb,
                                                 const float* __restrict__ embf,
                                                 const float* __restrict__ snorm,
                                                 float* __restrict__ out_idx,
                                                 __bf16* __restrict__ qb,
                                                 float* __restrict__ acc) {
  __shared__ __align__(16) __bf16 s_emb[2][VQ_GELEMS];   // 2 x 16 KB
  __shared__ int   s_win[VQ_RPB];
  __shared__ float s_red[VQ_THREADS];

  const int wid  = threadIdx.x >> 5;
  const int lane = threadIdx.x & 31;
  const int n    = lane & 15;
  const int rowB = blockIdx.x * VQ_RPB;    // block row base
  const int row0 = rowB + wid * VQ_RPW;    // this wave's rows (2 tiles of 16)

  // z operand tiles (D=128 -> 4 chunks per row-tile), loaded once
  v16bf a0[4], a1[4];
  #pragma unroll
  for (int kk = 0; kk < 4; ++kk) {
    a0[kk] = load_tile16x32(zb + (size_t)row0 * D_N + kk * 32, D_N);
    a1[kk] = load_tile16x32(zb + (size_t)(row0 + 16) * D_N + kk * 32, D_N);
  }

  float best0[8], best1[8];
  int   bidx0[8], bidx1[8];
  #pragma unroll
  for (int v = 0; v < 8; ++v) {
    best0[v] = -3.4e38f; bidx0[v] = 0;
    best1[v] = -3.4e38f; bidx1[v] = 0;
  }

  vq_stage_group(embb, s_emb[0], 0);
  vq_stage_wait();

  for (int g = 0; g < VQ_NG; ++g) {
    if (g + 1 < VQ_NG) vq_stage_group(embb, s_emb[(g + 1) & 1], g + 1);
    const __bf16* sbuf = s_emb[g & 1];

    #pragma unroll
    for (int t = 0; t < VQ_TG; ++t) {
      const int code = g * VQ_GCODES + t * 16 + n;   // this lane's column/code
      const float ci = -snorm[code];
      v8f c0 = {ci, ci, ci, ci, ci, ci, ci, ci};
      v8f c1 = c0;
      const __bf16* tb = sbuf + (t * 16) * D_N;
      v16bf b = load_tile16x32(tb, D_N);
      #pragma unroll
      for (int kk = 0; kk < 4; ++kk) {
        v16bf bn;
        if (kk + 1 < 4) bn = load_tile16x32(tb + (kk + 1) * 32, D_N);
        c0 = __builtin_amdgcn_wmma_f32_16x16x32_bf16(false, a0[kk], false, b,
                                                     (short)0, c0, false, false);
        c1 = __builtin_amdgcn_wmma_f32_16x16x32_bf16(false, a1[kk], false, b,
                                                     (short)0, c1, false, false);
        if (kk + 1 < 4) b = bn;
      }
      // Within a lane codes ascend, so strictly-greater keeps the smallest
      // index among ties (jnp.argmin first-occurrence); no eq-compare needed.
      #pragma unroll
      for (int v = 0; v < 8; ++v) {
        if (c0[v] > best0[v]) { best0[v] = c0[v]; bidx0[v] = code; }
        if (c1[v] > best1[v]) { best1[v] = c1[v]; bidx1[v] = code; }
      }
    }
    if (g + 1 < VQ_NG) vq_stage_wait();
  }

  // reduce over the 16 lanes that share each row (xor masks < 16 stay
  // in-half); tie-break on smaller code needed here (lanes hold diff codes)
  #pragma unroll
  for (int off = 8; off >= 1; off >>= 1) {
    #pragma unroll
    for (int v = 0; v < 8; ++v) {
      float os0 = __shfl_xor(best0[v], off);
      int   oi0 = __shfl_xor(bidx0[v], off);
      if (os0 > best0[v] || (os0 == best0[v] && oi0 < bidx0[v])) { best0[v] = os0; bidx0[v] = oi0; }
      float os1 = __shfl_xor(best1[v], off);
      int   oi1 = __shfl_xor(bidx1[v], off);
      if (os1 > best1[v] || (os1 == best1[v] && oi1 < bidx1[v])) { best1[v] = os1; bidx1[v] = oi1; }
    }
  }
  if (n == 0) {              // lane 0 -> rows m 0..7, lane 16 -> rows 8..15
    const int mb0 = wid * VQ_RPW + (lane >> 4) * 8;
    #pragma unroll
    for (int v = 0; v < 8; ++v) {
      s_win[mb0 + v] = bidx0[v];
      out_idx[rowB + mb0 + v] = (float)bidx0[v];     // idx emitted as float
      s_win[mb0 + 16 + v] = bidx1[v];
      out_idx[rowB + mb0 + 16 + v] = (float)bidx1[v];
    }
  }
  __syncthreads();

  // gather q (fp32), accumulate sum (z-q)^2, emit bf16 q for the decoder
  float local = 0.f;
  for (int e = threadIdx.x; e < VQ_RPB * D_N; e += VQ_THREADS) {
    const int r = e >> 7, d = e & 127;
    const int ci = s_win[r];
    const float qv = embf[(size_t)ci * D_N + d];
    const float zv = zf[(size_t)(rowB + r) * D_N + d];
    const float df = zv - qv;
    local += df * df;
    qb[(size_t)(rowB + r) * D_N + d] = (__bf16)qv;
  }
  s_red[threadIdx.x] = local;
  __syncthreads();
  for (int s = VQ_THREADS / 2; s > 0; s >>= 1) {
    if (threadIdx.x < s) s_red[threadIdx.x] += s_red[threadIdx.x + s];
    __syncthreads();
  }
  if (threadIdx.x == 0) atomicAdd(acc + 0, s_red[0]);
}

// ---------------- reconstruction head (K=256 -> 6, tanh) + loss ----------

__global__ __launch_bounds__(256) void head_kernel(const __bf16* __restrict__ d2,
                                                   const float* __restrict__ Wh,
                                                   const float* __restrict__ bh,
                                                   const float* __restrict__ action,
                                                   float* __restrict__ out_rec,
                                                   float* __restrict__ acc) {
  __shared__ float s_red[256];
  const int g   = blockIdx.x * 256 + threadIdx.x;  // B*A threads exactly
  const int row = g / A_N;
  const int col = g % A_N;
  float s = bh[col];
  const __bf16* dr = d2 + (size_t)row * H_N;
  #pragma unroll 8
  for (int k = 0; k < H_N; ++k) s += (float)dr[k] * Wh[k * A_N + col];
  const float r = tanhf(s);
  out_rec[g] = r;
  const float df = r - action[g];
  s_red[threadIdx.x] = df * df;
  __syncthreads();
  for (int t = 128; t > 0; t >>= 1) {
    if (threadIdx.x < t) s_red[threadIdx.x] += s_red[threadIdx.x + t];
    __syncthreads();
  }
  if (threadIdx.x == 0) atomicAdd(acc + 1, s_red[0]);
}

__global__ void final_kernel(const float* __restrict__ acc, float* __restrict__ out) {
  const float vq_mse  = acc[0] / (float)((size_t)B_N * D_N);
  const float vq_loss = 1.25f * vq_mse;            // beta*commit + embed, equal fwd
  const float rec     = acc[1] / (float)((size_t)B_N * A_N);
  out[0] = rec + vq_loss;
  out[1] = vq_loss;
  out[2] = rec;
}

// ---------------------------------------------------------------------------

extern "C" void kernel_launch(void* const* d_in, const int* in_sizes, int n_in,
                              void* d_out, int out_size, void* d_ws, size_t ws_size,
                              hipStream_t stream) {
  const float* action = (const float*)d_in[0];
  const float* We1 = (const float*)d_in[1];
  const float* be1 = (const float*)d_in[2];
  const float* We2 = (const float*)d_in[3];
  const float* be2 = (const float*)d_in[4];
  const float* We3 = (const float*)d_in[5];
  const float* be3 = (const float*)d_in[6];
  const float* emb = (const float*)d_in[7];
  const float* Wd1 = (const float*)d_in[8];
  const float* bd1 = (const float*)d_in[9];
  const float* Wd2 = (const float*)d_in[10];
  const float* bd2 = (const float*)d_in[11];
  const float* Wh  = (const float*)d_in[12];
  const float* bh  = (const float*)d_in[13];
  float* out = (float*)d_out;
  float* out_idx = out + 3;
  float* out_rec = out + 3 + B_N;

  char* ws = (char*)d_ws;
  size_t off = 0;
  auto alloc = [&](size_t bytes) -> char* {
    char* p = ws + off;
    off = (off + bytes + 255) & ~(size_t)255;
    return p;
  };
  __bf16* We2t  = (__bf16*)alloc((size_t)H_N * H_N * 2);   // [n=256][k=256]
  __bf16* We3t  = (__bf16*)alloc((size_t)D_N * H_N * 2);   // [n=128][k=256]
  __bf16* Wd1t  = (__bf16*)alloc((size_t)H_N * D_N * 2);   // [n=256][k=128]
  __bf16* Wd2t  = (__bf16*)alloc((size_t)H_N * H_N * 2);   // [n=256][k=256]
  __bf16* embb  = (__bf16*)alloc((size_t)K_N * D_N * 2);
  float*  snorm = (float*)alloc((size_t)K_N * 4);
  float*  acc   = (float*)alloc(256);
  __bf16* h1    = (__bf16*)alloc((size_t)B_N * H_N * 2);
  __bf16* h2    = (__bf16*)alloc((size_t)B_N * H_N * 2);
  float*  zf    = (float*)alloc((size_t)B_N * D_N * 4);
  __bf16* zb    = (__bf16*)alloc((size_t)B_N * D_N * 2);
  __bf16* qb    = (__bf16*)alloc((size_t)B_N * D_N * 2);
  __bf16* d1b   = (__bf16*)alloc((size_t)B_N * H_N * 2);
  __bf16* d2b   = (__bf16*)alloc((size_t)B_N * H_N * 2);
  (void)ws_size; (void)in_sizes; (void)n_in; (void)out_size;

  hipMemsetAsync(acc, 0, 2 * sizeof(float), stream);

  // weight prep
  tconv_kernel<<<(H_N * H_N + 255) / 256, 256, 0, stream>>>(We2, We2t, H_N, H_N);
  tconv_kernel<<<(H_N * D_N + 255) / 256, 256, 0, stream>>>(We3, We3t, H_N, D_N);
  tconv_kernel<<<(D_N * H_N + 255) / 256, 256, 0, stream>>>(Wd1, Wd1t, D_N, H_N);
  tconv_kernel<<<(H_N * H_N + 255) / 256, 256, 0, stream>>>(Wd2, Wd2t, H_N, H_N);
  conv_kernel<<<(K_N * D_N + 255) / 256, 256, 0, stream>>>(emb, embb, K_N * D_N);
  snorm_kernel<<<(K_N + 255) / 256, 256, 0, stream>>>(emb, snorm);

  // encoder
  enc1_kernel<<<(B_N * H_N) / 256, 256, 0, stream>>>(action, We1, be1, h1);
  gemm_bf16_wmma<H_N, true, false><<<dim3(B_N / 32, H_N / 64), 256, 0, stream>>>(
      h1, We2t, be2, h2, nullptr, H_N);
  gemm_bf16_wmma<H_N, false, true><<<dim3(B_N / 32, D_N / 64), 256, 0, stream>>>(
      h2, We3t, be3, zb, zf, D_N);

  // vector quantizer (fused distances + argmin + gather + vq loss)
  vq_kernel<<<B_N / VQ_RPB, VQ_THREADS, 0, stream>>>(zb, zf, embb, emb, snorm,
                                                     out_idx, qb, acc);

  // decoder + head
  gemm_bf16_wmma<D_N, true, false><<<dim3(B_N / 32, H_N / 64), 256, 0, stream>>>(
      qb, Wd1t, bd1, d1b, nullptr, H_N);
  gemm_bf16_wmma<H_N, true, false><<<dim3(B_N / 32, H_N / 64), 256, 0, stream>>>(
      d1b, Wd2t, bd2, d2b, nullptr, H_N);
  head_kernel<<<(B_N * A_N) / 256, 256, 0, stream>>>(d2b, Wh, bh, action, out_rec, acc);

  final_kernel<<<1, 1, 0, stream>>>(acc, out);
}
